// HGT_52974126629059
// MI455X (gfx1250) — matrix operand order repaired and checked by reading, not compiled, not run on hardware
//
// HGT forward for MI455X (gfx1250, wave32, WMMA).
// Dense GEMMs run on v_wmma_f32_16x16x32_bf16 (f32 accumulate) with the B
// operand staged once per block into LDS in WMMA-fragment order (8x less
// redundant global weight traffic, contiguous ds_load_b128 feeds). Graph
// scatter/softmax parts are bandwidth-bound VALU kernels with float atomics
// (entire working set fits the 192MB L2). Workspace arena ~257MB.
#include <hip/hip_runtime.h>
#include <math.h>

typedef __bf16 bf16;
typedef __bf16 bf16x16 __attribute__((ext_vector_type(16)));
typedef float  f32x8   __attribute__((ext_vector_type(8)));

#define N_OPC   20000
#define N_VALC  20000
#define NNODE   40000
#define EDGES   160000
#define HIDC    256
#define NODE_ELEMS (20000 * 256)
#define CDIV(a, b) (((a) + (b) - 1) / (b))

// ---------------- device helpers ----------------
__device__ __forceinline__ float geluf(float x) {
  return 0.5f * x * (1.0f + erff(x * 0.7071067811865476f));
}
// monotonic float <-> uint mapping for atomic max on floats
__device__ __forceinline__ unsigned fkey(float f) {
  unsigned b = __float_as_uint(f);
  return (b & 0x80000000u) ? ~b : (b | 0x80000000u);
}
__device__ __forceinline__ float funkey(unsigned k) {
  unsigned b = (k & 0x80000000u) ? (k & 0x7fffffffu) : ~k;
  return __uint_as_float(b);
}

// ---------------- WMMA GEMM: C[M,256] = A[M,K](bf16) @ W[K,256](f32) + bias
// Block = 8 waves covering 8 M-tiles x 4 N-tiles. The K x 64 weight slab is
// converted to bf16 and staged in LDS in exact B-fragment order:
//   lb[((ks*4 + nt)*32 + lane)*16 + i] = W[ks*32 + (lane>>4)*16 + i, tnb + nt*16 + (lane&15)]
// so each wave pulls a fragment with two contiguous ds_load_b128.
// Optional epilogue: skip blend  C = sig(g)*C + (1-sig(g))*Xres
__global__ __launch_bounds__(256)
void k_gemm_wmma(const bf16* __restrict__ A, const float* __restrict__ W,
                 const float* __restrict__ bias, int M, int K,
                 float* __restrict__ Cf, bf16* __restrict__ Cb,
                 const float* __restrict__ skip_gate, const float* __restrict__ Xres) {
  const int N = 256;
  __shared__ bf16 lb[8 * 4 * 32 * 16];          // up to K=256: 32KB
  int mtiles = M >> 4;
  int mb = blockIdx.x >> 2;                     // M super-block (8 tiles)
  int ng = blockIdx.x & 3;                      // N group (4 tiles = 64 cols)
  int tnb = ng << 6;
  int nsteps = K >> 5;

  // ---- cooperative B staging (all waves participate) ----
  int entries = nsteps * 4 * 32;
  for (int ent = threadIdx.x; ent < entries; ent += 256) {
    int lane_e = ent & 31;
    int nt = (ent >> 5) & 3;
    int ks = ent >> 7;
    int kh = lane_e >> 4;
    const float* wp = W + (size_t)(ks * 32 + kh * 16) * N + tnb + nt * 16 + (lane_e & 15);
    bf16* dst = lb + (size_t)ent * 16;
#pragma unroll
    for (int i = 0; i < 16; ++i) dst[i] = (bf16)wp[(size_t)i * N];
  }
  __syncthreads();

  // ---- per-wave compute: 1 M-tile x 4 N-tiles ----
  int wv = threadIdx.x >> 5;
  int lane = threadIdx.x & 31;
  int mt = mb * 8 + wv;
  if (mt >= mtiles) return;                     // wave-uniform exit after barrier
  int tm = mt << 4;
  int m  = lane & 15;                           // row (A) / col (B,C) within tile
  int kh = lane >> 4;
  f32x8 z = {0.f, 0.f, 0.f, 0.f, 0.f, 0.f, 0.f, 0.f};
  f32x8 acc[4] = {z, z, z, z};
  const bf16* arow0 = A + (size_t)(tm + m) * K + kh * 16;
  for (int ks = 0; ks < nsteps; ++ks) {
    bf16x16 a;
    const bf16* arow = arow0 + ks * 32;
#pragma unroll
    for (int i = 0; i < 8; ++i) a[i] = arow[i];
#pragma unroll
    for (int i = 0; i < 8; ++i) a[8 + i] = arow[8 + i];
    if (ks + 1 < nsteps) __builtin_prefetch(arow + 32, 0, 0);  // global_prefetch_b8
#pragma unroll
    for (int nt = 0; nt < 4; ++nt) {
      const bf16* bp = lb + (size_t)(((ks * 4 + nt) * 32 + lane)) * 16;
      bf16x16 b;
#pragma unroll
      for (int i = 0; i < 16; ++i) b[i] = bp[i];
      acc[nt] = __builtin_amdgcn_wmma_f32_16x16x32_bf16(false, a, false, b,
                                                        (short)0, acc[nt], false, false);
    }
  }
  float sk = 1.0f, sk1 = 0.0f;
  if (skip_gate) { float g = 1.0f / (1.0f + __expf(-skip_gate[0])); sk = g; sk1 = 1.0f - g; }
#pragma unroll
  for (int nt = 0; nt < 4; ++nt) {
    int tn = tnb + nt * 16;
#pragma unroll
    for (int r = 0; r < 8; ++r) {
      int mm = tm + r + 8 * kh;                 // C/D VGPR layout: M=r | M=8+r
      size_t o = (size_t)mm * N + tn + m;
      float cv = acc[nt][r];
      if (bias) cv += bias[tn + m];
      if (skip_gate) cv = sk * cv + sk1 * Xres[o];
      if (Cf) Cf[o] = cv;
      if (Cb) Cb[o] = (bf16)cv;
    }
  }
}

// Block-diagonal per-head transform: C[n, h*64+e] = sum_d A[n,h*64+d] * Wr[h,d,e]
// (Wr is only 64KB total and L2/L1-resident; direct global feed is fine here.)
__global__ __launch_bounds__(256)
void k_gemm_rel(const bf16* __restrict__ A, const float* __restrict__ Wr,
                float* __restrict__ Cf, int M) {
  int wave = blockIdx.x * (blockDim.x >> 5) + (threadIdx.x >> 5);
  int lane = threadIdx.x & 31;
  int tiles = (M >> 4) << 4;
  if (wave >= tiles) return;
  int tm = (wave >> 4) << 4;
  int tn = (wave & 15) << 4;
  int h  = tn >> 6;
  int e0 = tn & 63;
  int m  = lane & 15;
  int kh = lane >> 4;
  f32x8 acc = {0.f, 0.f, 0.f, 0.f, 0.f, 0.f, 0.f, 0.f};
  for (int kk = 0; kk < 64; kk += 32) {
    bf16x16 a, b;
    const bf16* arow = A + (size_t)(tm + m) * 256 + h * 64 + kk + kh * 16;
#pragma unroll
    for (int i = 0; i < 8; ++i) a[i] = arow[i];
#pragma unroll
    for (int i = 0; i < 8; ++i) a[8 + i] = arow[8 + i];
    const float* wcol = Wr + h * 4096 + (size_t)(kk + kh * 16) * 64 + e0 + m;
#pragma unroll
    for (int i = 0; i < 16; ++i) b[i] = (bf16)wcol[(size_t)i * 64];
    acc = __builtin_amdgcn_wmma_f32_16x16x32_bf16(false, a, false, b,
                                                  (short)0, acc, false, false);
  }
#pragma unroll
  for (int r = 0; r < 8; ++r) {
    int mm = tm + r + 8 * kh;
    Cf[(size_t)mm * 256 + tn + m] = acc[r];
  }
}

// ---------------- elementwise / conversion ----------------
__global__ void k_f32_to_bf(const float* __restrict__ in, bf16* __restrict__ out, int n) {
  int i = blockIdx.x * blockDim.x + threadIdx.x;
  if (i < n) out[i] = (bf16)in[i];
}
__global__ void k_gelu2bf(const float* __restrict__ in, bf16* __restrict__ out, int n) {
  int i = blockIdx.x * blockDim.x + threadIdx.x;
  if (i < n) out[i] = (bf16)geluf(in[i]);
}
__global__ void k_gelu_f32(const float* __restrict__ in, float* __restrict__ out, int n) {
  int i = blockIdx.x * blockDim.x + threadIdx.x;
  if (i < n) out[i] = geluf(in[i]);
}
__global__ void k_zero(float* __restrict__ p, long long n) {
  long long stride = (long long)gridDim.x * blockDim.x;
  for (long long i = blockIdx.x * (long long)blockDim.x + threadIdx.x; i < n; i += stride)
    p[i] = 0.0f;
}

// ---------------- HGT edge stage ----------------
__global__ void k_seg_init(unsigned* __restrict__ key, float* __restrict__ sum, int n) {
  int i = blockIdx.x * blockDim.x + threadIdx.x;
  if (i < n) { key[i] = fkey(-INFINITY); sum[i] = 0.0f; }
}
__global__ void k_edge_logits(const int* __restrict__ ei, const float* __restrict__ Q,
                              const float* __restrict__ KR, const float* __restrict__ prel,
                              float* __restrict__ logit, unsigned* __restrict__ segkey) {
  int idx = blockIdx.x * blockDim.x + threadIdx.x;
  if (idx >= EDGES * 4) return;
  int e = idx >> 2, h = idx & 3;
  int s = ei[e], d = ei[EDGES + e];
  const float* q = Q + (size_t)d * 256 + h * 64;
  const float* k = KR + (size_t)s * 256 + h * 64;
  float acc = 0.f;
#pragma unroll 8
  for (int i = 0; i < 64; ++i) acc += q[i] * k[i];
  float l = acc * prel[h] * 0.125f;       // / sqrt(DH=64)
  logit[idx] = l;
  atomicMax(&segkey[d * 4 + h], fkey(l));
}
__global__ void k_edge_exp(const int* __restrict__ ei, float* __restrict__ logit,
                           const unsigned* __restrict__ segkey, float* __restrict__ segsum) {
  int idx = blockIdx.x * blockDim.x + threadIdx.x;
  if (idx >= EDGES * 4) return;
  int e = idx >> 2, h = idx & 3;
  int d = ei[EDGES + e];
  float ev = __expf(logit[idx] - funkey(segkey[d * 4 + h]));
  logit[idx] = ev;
  atomicAdd(&segsum[d * 4 + h], ev);
}
__global__ void k_edge_agg(const int* __restrict__ ei, const float* __restrict__ elog,
                           const float* __restrict__ segsum, const float* __restrict__ VR,
                           float* __restrict__ agg) {
  int idx = blockIdx.x * blockDim.x + threadIdx.x;
  if (idx >= EDGES * 64) return;
  int e = idx >> 6, t = idx & 63;
  int c = t * 4, h = t >> 4;
  int s = ei[e], d = ei[EDGES + e];
  float a = elog[e * 4 + h] / (segsum[d * 4 + h] + 1e-16f);
  const float* v = VR + (size_t)s * 256 + c;
  float* o = agg + (size_t)d * 256 + c;
  atomicAdd(o + 0, v[0] * a);
  atomicAdd(o + 1, v[1] * a);
  atomicAdd(o + 2, v[2] * a);
  atomicAdd(o + 3, v[3] * a);
}

// ---------------- graph norm (deterministic two-stage) ----------------
__global__ void k_reduce_part(const float* __restrict__ x, long long n,
                              float* __restrict__ ps, float* __restrict__ pq) {
  __shared__ float ss[256], sq[256];
  float s = 0.f, q = 0.f;
  long long stride = (long long)gridDim.x * blockDim.x;
  for (long long i = blockIdx.x * (long long)blockDim.x + threadIdx.x; i < n; i += stride) {
    float v = x[i]; s += v; q += v * v;
  }
  ss[threadIdx.x] = s; sq[threadIdx.x] = q; __syncthreads();
  for (int w = 128; w > 0; w >>= 1) {
    if (threadIdx.x < w) { ss[threadIdx.x] += ss[threadIdx.x + w]; sq[threadIdx.x] += sq[threadIdx.x + w]; }
    __syncthreads();
  }
  if (threadIdx.x == 0) { ps[blockIdx.x] = ss[0]; pq[blockIdx.x] = sq[0]; }
}
__global__ void k_reduce_fin(const float* __restrict__ ps, const float* __restrict__ pq,
                             float count, float* __restrict__ stats) {
  __shared__ float ss[256], sq[256];
  float s = 0.f, q = 0.f;
  for (int i = threadIdx.x; i < 1024; i += 256) { s += ps[i]; q += pq[i]; }
  ss[threadIdx.x] = s; sq[threadIdx.x] = q; __syncthreads();
  for (int w = 128; w > 0; w >>= 1) {
    if (threadIdx.x < w) { ss[threadIdx.x] += ss[threadIdx.x + w]; sq[threadIdx.x] += sq[threadIdx.x + w]; }
    __syncthreads();
  }
  if (threadIdx.x == 0) {
    float mu = ss[0] / count;
    float var = sq[0] / count - mu * mu;
    stats[0] = mu;
    stats[1] = rsqrtf(var + 1e-5f);
  }
}
__global__ void k_norm_apply(const float* __restrict__ in, const float* __restrict__ stats,
                             const float* __restrict__ w, const float* __restrict__ b,
                             float* __restrict__ outF, bf16* __restrict__ outB, int n) {
  int i = blockIdx.x * blockDim.x + threadIdx.x;
  if (i >= n) return;
  int c = i & 255;
  float g = geluf((in[i] - stats[0]) * stats[1] * w[c] + b[c]);
  outF[i] = g;
  outB[i] = (bf16)g;
}

// ---------------- final pooling stage ----------------
__global__ void k_concat(const float* __restrict__ xop, const float* __restrict__ xval,
                         float* __restrict__ xh, int n) {
  int i = blockIdx.x * blockDim.x + threadIdx.x;
  if (i >= n) return;
  int node = i >> 8;
  xh[i] = (node < N_OPC) ? xop[i] : xval[i - N_OPC * 256];
}
__global__ void k_gat_node(const float* __restrict__ xh, const float* __restrict__ gw,
                           const float* __restrict__ asrc, const float* __restrict__ adst,
                           float* __restrict__ xp, float* __restrict__ as_,
                           float* __restrict__ ad_, unsigned* __restrict__ gkey,
                           float* __restrict__ gsum, float* __restrict__ score) {
  int n = blockIdx.x * blockDim.x + threadIdx.x;
  if (n >= NNODE) return;
  const float* x = xh + (size_t)n * 256;
  float acc = 0.f;
#pragma unroll 8
  for (int c = 0; c < 256; ++c) acc += x[c] * gw[c];
  xp[n] = acc;
  as_[n] = acc * asrc[0];
  ad_[n] = acc * adst[0];
  gkey[n] = fkey(-INFINITY);
  gsum[n] = 0.f;
  score[n] = 0.f;
}
__device__ __forceinline__ void gat_decode(int i, const int* ov, const int* vo, int* s, int* d) {
  if (i < EDGES)          { *s = ov[i];               *d = ov[EDGES + i] + N_OPC; }
  else if (i < 2 * EDGES) { int j = i - EDGES; *s = vo[j] + N_OPC; *d = vo[EDGES + j]; }
  else                    { *s = *d = i - 2 * EDGES; }
}
__global__ void k_gat_logit(const int* __restrict__ ov, const int* __restrict__ vo,
                            const float* __restrict__ as_, const float* __restrict__ ad_,
                            float* __restrict__ elog, unsigned* __restrict__ gkey, int eg) {
  int i = blockIdx.x * blockDim.x + threadIdx.x;
  if (i >= eg) return;
  int s, d; gat_decode(i, ov, vo, &s, &d);
  float l = as_[s] + ad_[d];
  l = (l > 0.f) ? l : 0.2f * l;           // leaky_relu 0.2
  elog[i] = l;
  atomicMax(&gkey[d], fkey(l));
}
__global__ void k_gat_exp(const int* __restrict__ ov, const int* __restrict__ vo,
                          float* __restrict__ elog, const unsigned* __restrict__ gkey,
                          float* __restrict__ gsum, int eg) {
  int i = blockIdx.x * blockDim.x + threadIdx.x;
  if (i >= eg) return;
  int s, d; gat_decode(i, ov, vo, &s, &d);
  float ev = __expf(elog[i] - funkey(gkey[d]));
  elog[i] = ev;
  atomicAdd(&gsum[d], ev);
}
__global__ void k_gat_score(const int* __restrict__ ov, const int* __restrict__ vo,
                            const float* __restrict__ elog, const float* __restrict__ gsum,
                            const float* __restrict__ xp, float* __restrict__ score, int eg) {
  int i = blockIdx.x * blockDim.x + threadIdx.x;
  if (i >= eg) return;
  int s, d; gat_decode(i, ov, vo, &s, &d);
  float a = elog[i] / (gsum[d] + 1e-16f);
  atomicAdd(&score[d], a * xp[s]);
}
__global__ void k_gat_tanh(float* __restrict__ score, const float* __restrict__ bias) {
  int n = blockIdx.x * blockDim.x + threadIdx.x;
  if (n < NNODE) score[n] = tanhf(score[n] + bias[0]);
}
// single-block top-16 with jax.lax.top_k tie rule (smaller index wins)
__global__ void k_topk16(float* __restrict__ score, int n,
                         float* __restrict__ topv, int* __restrict__ topi) {
  __shared__ float sv[1024];
  __shared__ int   si[1024];
  for (int it = 0; it < 16; ++it) {
    float bv = -INFINITY; int bi = 0x7fffffff;
    for (int i = threadIdx.x; i < n; i += blockDim.x) {
      float v = score[i];
      if (v > bv || (v == bv && i < bi)) { bv = v; bi = i; }
    }
    sv[threadIdx.x] = bv; si[threadIdx.x] = bi; __syncthreads();
    for (int w = 512; w > 0; w >>= 1) {
      if (threadIdx.x < w) {
        float ov = sv[threadIdx.x + w]; int oi = si[threadIdx.x + w];
        if (ov > sv[threadIdx.x] || (ov == sv[threadIdx.x] && oi < si[threadIdx.x])) {
          sv[threadIdx.x] = ov; si[threadIdx.x] = oi;
        }
      }
      __syncthreads();
    }
    if (threadIdx.x == 0) { topv[it] = sv[0]; topi[it] = si[0]; score[si[0]] = -INFINITY; }
    __syncthreads();
  }
}
// single-block pooled readout + MLP head
__global__ void k_head(const float* __restrict__ xh, const float* __restrict__ topv,
                       const int* __restrict__ topi,
                       const float* r1b, const float* r1w,
                       const float* r2b, const float* r2w,
                       const float* m1b, const float* m1w,
                       const float* m2b, const float* m2w,
                       const float* m3b, const float* m3w,
                       const float* m4b, const float* m4w,
                       float* __restrict__ out) {
  __shared__ float pool[4096];
  __shared__ float bufA[256];
  __shared__ float bufB[256];
  int t = threadIdx.x;
  for (int j = t; j < 4096; j += 256) {
    int r = j >> 8, c = j & 255;
    pool[j] = xh[(size_t)topi[r] * 256 + c] * topv[r];
  }
  __syncthreads();
  { // readout l1: 4096 -> 256, gelu
    float acc = r1b[t];
    for (int j = 0; j < 4096; ++j) acc += pool[j] * r1w[(size_t)j * 256 + t];
    bufA[t] = geluf(acc);
  }
  __syncthreads();
  { // readout l2: 256 -> 256 (x_agg)
    float acc = r2b[t];
    for (int j = 0; j < 256; ++j) acc += bufA[j] * r2w[j * 256 + t];
    bufB[t] = acc;
  }
  __syncthreads();
  { // mlp l1: 256 -> 256, gelu
    float acc = m1b[t];
    for (int j = 0; j < 256; ++j) acc += bufB[j] * m1w[j * 256 + t];
    bufA[t] = geluf(acc);
  }
  __syncthreads();
  if (t < 128) { // mlp l2: 256 -> 128, gelu
    float acc = m2b[t];
    for (int j = 0; j < 256; ++j) acc += bufA[j] * m2w[j * 128 + t];
    bufB[t] = geluf(acc);
  }
  __syncthreads();
  if (t < 64) { // mlp l3: 128 -> 64, gelu
    float acc = m3b[t];
    for (int j = 0; j < 128; ++j) acc += bufB[j] * m3w[j * 64 + t];
    bufA[t] = geluf(acc);
  }
  __syncthreads();
  if (t == 0) { // mlp l4: 64 -> 1, nan_to_num
    float acc = m4b[0];
    for (int j = 0; j < 64; ++j) acc += bufA[j] * m4w[j];
    if (acc != acc) acc = 0.0f;
    else if (acc > 3.4028234663852886e38f) acc = 3.4028234663852886e38f;
    else if (acc < -3.4028234663852886e38f) acc = -3.4028234663852886e38f;
    out[0] = acc;
  }
}

// ---------------- workspace layout (bytes) ----------------
static const size_t NF4 = (size_t)NODE_ELEMS * 4;   // 20,480,000
static const size_t NB2 = (size_t)NODE_ELEMS * 2;   // 10,240,000
static const size_t OFF_XOPF   = 0;
static const size_t OFF_XVALF  = OFF_XOPF  + NF4;
static const size_t OFF_OUTOPF = OFF_XVALF + NF4;
static const size_t OFF_OUTVAF = OFF_OUTOPF + NF4;
static const size_t OFF_QOP    = OFF_OUTVAF + NF4;   // QOP+QVAL contiguous -> xh alias
static const size_t OFF_QVAL   = OFF_QOP   + NF4;
static const size_t OFF_KROV   = OFF_QVAL  + NF4;    // aliased by GAT scalars later
static const size_t OFF_VROV   = OFF_KROV  + NF4;
static const size_t OFF_KRVO   = OFF_VROV  + NF4;
static const size_t OFF_VRVO   = OFF_KRVO  + NF4;
static const size_t OFF_AGGF   = OFF_VRVO  + NF4;
static const size_t OFF_XOPB   = OFF_AGGF  + NF4;
static const size_t OFF_XVALB  = OFF_XOPB  + NB2;
static const size_t OFF_TMPB   = OFF_XVALB + NB2;
static const size_t OFF_AGGB   = OFF_TMPB  + NB2;
static const size_t OFF_LOGIT  = OFF_AGGB  + NB2;                 // 640000 floats
static const size_t OFF_SEGKEY = OFF_LOGIT + 640000ull * 4;
static const size_t OFF_SEGSUM = OFF_SEGKEY + 80000ull * 4;
static const size_t OFF_REDP   = OFF_SEGSUM + 80000ull * 4;
static const size_t OFF_REDQ   = OFF_REDP + 1024 * 4;
static const size_t OFF_STATS  = OFF_REDQ + 1024 * 4;
static const size_t OFF_TOPV   = OFF_STATS + 256;
static const size_t OFF_TOPI   = OFF_TOPV + 64;
static const size_t WS_NEEDED  = OFF_TOPI + 64;

extern "C" void kernel_launch(void* const* d_in, const int* in_sizes, int n_in,
                              void* d_out, int out_size, void* d_ws, size_t ws_size,
                              hipStream_t stream) {
  if (n_in < 100 || ws_size < WS_NEEDED) return;
  auto F = [&](int i) { return (const float*)d_in[i]; };
  const float* x_op  = F(0);
  const float* x_val = F(1);
  const int*   ei_ov = (const int*)d_in[2];
  const int*   ei_vo = (const int*)d_in[3];
  // params (jax pytree order: dict keys sorted)
  const float *gat_ad = F(4), *gat_as = F(5), *gat_b = F(6), *gat_w = F(7);
  const float *m1b = F(80), *m1w = F(81), *m2b = F(82), *m2w = F(83);
  const float *m3b = F(84), *m3w = F(85), *m4b = F(86), *m4w = F(87);
  const float *norm_b[2][2] = {{F(88), F(90)}, {F(92), F(94)}};  // [op/val][layer]
  const float *norm_w[2][2] = {{F(89), F(91)}, {F(93), F(95)}};
  const float *r1b = F(96), *r1w = F(97), *r2b = F(98), *r2w = F(99);

  char* w = (char*)d_ws;
  float* XOPF  = (float*)(w + OFF_XOPF);
  float* XVALF = (float*)(w + OFF_XVALF);
  float* OUTOP = (float*)(w + OFF_OUTOPF);
  float* OUTVA = (float*)(w + OFF_OUTVAF);
  float* QOP   = (float*)(w + OFF_QOP);
  float* QVAL  = (float*)(w + OFF_QVAL);
  float* KROV  = (float*)(w + OFF_KROV);
  float* VROV  = (float*)(w + OFF_VROV);
  float* KRVO  = (float*)(w + OFF_KRVO);
  float* VRVO  = (float*)(w + OFF_VRVO);
  float* AGGF  = (float*)(w + OFF_AGGF);
  bf16*  XOPB  = (bf16*)(w + OFF_XOPB);
  bf16*  XVALB = (bf16*)(w + OFF_XVALB);
  bf16*  TMPB  = (bf16*)(w + OFF_TMPB);
  bf16*  AGGB  = (bf16*)(w + OFF_AGGB);
  float* LOGIT = (float*)(w + OFF_LOGIT);
  unsigned* SEGKEY = (unsigned*)(w + OFF_SEGKEY);
  float* SEGSUM = (float*)(w + OFF_SEGSUM);
  float* REDP  = (float*)(w + OFF_REDP);
  float* REDQ  = (float*)(w + OFF_REDQ);
  float* STATS = (float*)(w + OFF_STATS);
  float* TOPV  = (float*)(w + OFF_TOPV);
  int*   TOPI  = (int*)(w + OFF_TOPI);
  // final-stage aliases
  float* XH   = QOP;                        // 40000x256 spans QOP+QVAL
  float* XP   = KROV;                       // GAT scalars live in freed KROV
  float* AS_  = KROV + 40000;
  float* AD_  = KROV + 80000;
  unsigned* GKEY = (unsigned*)(KROV + 120000);
  float* GSUM = KROV + 160000;
  float* SCORE = KROV + 200000;
  float* ELOG = LOGIT;                      // 360000 <= 640000

  const int GEMM_BLOCKS = CDIV(1250, 8) * 4;  // (Mtiles/8) x 4 N-groups = 628
  auto gemm = [&](const bf16* A, const float* W, const float* B, int K,
                  float* Cf, bf16* Cb, const float* skip, const float* Xres) {
    k_gemm_wmma<<<GEMM_BLOCKS, 256, 0, stream>>>(A, W, B, 20000, K, Cf, Cb, skip, Xres);
  };
  auto rel = [&](const bf16* A, const float* Wr, float* Cf) {
    k_gemm_rel<<<2500, 256, 0, stream>>>(A, Wr, Cf, 20000);
  };
  auto edge_stage = [&](const int* ei, int n_dst, const float* Q, const float* KR,
                        const float* VR, const float* prel, const float* ow,
                        const float* ob, const float* skip, const float* xres,
                        float* outF) {
    k_seg_init<<<CDIV(n_dst * 4, 256), 256, 0, stream>>>(SEGKEY, SEGSUM, n_dst * 4);
    k_edge_logits<<<CDIV(EDGES * 4, 256), 256, 0, stream>>>(ei, Q, KR, prel, LOGIT, SEGKEY);
    k_edge_exp<<<CDIV(EDGES * 4, 256), 256, 0, stream>>>(ei, LOGIT, SEGKEY, SEGSUM);
    k_zero<<<4096, 256, 0, stream>>>(AGGF, (long long)n_dst * 256);
    k_edge_agg<<<CDIV(EDGES * 64, 256), 256, 0, stream>>>(ei, LOGIT, SEGSUM, VR, AGGF);
    k_gelu2bf<<<CDIV(n_dst * 256, 256), 256, 0, stream>>>(AGGF, AGGB, n_dst * 256);
    k_gemm_wmma<<<CDIV(n_dst / 16, 8) * 4, 256, 0, stream>>>(
        AGGB, ow, ob, n_dst, 256, outF, nullptr, skip, xres);
  };
  auto norm_stage = [&](const float* inF, const float* nw, const float* nb,
                        float* xf, bf16* xb) {
    k_reduce_part<<<1024, 256, 0, stream>>>(inF, (long long)NODE_ELEMS, REDP, REDQ);
    k_reduce_fin<<<1, 256, 0, stream>>>(REDP, REDQ, (float)NODE_ELEMS, STATS);
    k_norm_apply<<<CDIV(NODE_ELEMS, 256), 256, 0, stream>>>(inF, STATS, nw, nb, xf, xb, NODE_ELEMS);
  };

  // inputs -> bf16 (layer 0 uses fin=128; skip path unused so no f32 copy needed)
  k_f32_to_bf<<<CDIV(20000 * 128, 256), 256, 0, stream>>>(x_op, XOPB, 20000 * 128);
  k_f32_to_bf<<<CDIV(20000 * 128, 256), 256, 0, stream>>>(x_val, XVALB, 20000 * 128);

  for (int L = 0; L < 3; ++L) {
    int fin = (L == 0) ? 128 : 256;
    int b = 8 + L * 24;
    const float *a_ov = F(b + 0), *a_vo = F(b + 1);
    const float *kb_op = F(b + 2), *kw_op = F(b + 3), *kb_val = F(b + 4), *kw_val = F(b + 5);
    const float *mm_ov = F(b + 6), *mm_vo = F(b + 7);
    const float *ob_op = F(b + 8), *ow_op = F(b + 9), *ob_val = F(b + 10), *ow_val = F(b + 11);
    const float *p_ov = F(b + 12), *p_vo = F(b + 13);
    const float *qb_op = F(b + 14), *qw_op = F(b + 15), *qb_val = F(b + 16), *qw_val = F(b + 17);
    const float *sk_op = F(b + 18), *sk_val = F(b + 19);
    const float *vb_op = F(b + 20), *vw_op = F(b + 21), *vb_val = F(b + 22), *vw_val = F(b + 23);

    // projections (WMMA)
    gemm(XVALB, qw_val, qb_val, fin, QVAL, nullptr, nullptr, nullptr);
    gemm(XOPB,  qw_op,  qb_op,  fin, QOP,  nullptr, nullptr, nullptr);
    gemm(XOPB,  kw_op,  kb_op,  fin, nullptr, TMPB, nullptr, nullptr);  rel(TMPB, a_ov,  KROV);
    gemm(XOPB,  vw_op,  vb_op,  fin, nullptr, TMPB, nullptr, nullptr);  rel(TMPB, mm_ov, VROV);
    gemm(XVALB, kw_val, kb_val, fin, nullptr, TMPB, nullptr, nullptr);  rel(TMPB, a_vo,  KRVO);
    gemm(XVALB, vw_val, vb_val, fin, nullptr, TMPB, nullptr, nullptr);  rel(TMPB, mm_vo, VRVO);

    // edge attention + aggregation + out projection (with skip for L>=1)
    edge_stage(ei_ov, N_VALC, QVAL, KROV, VROV, p_ov, ow_val, ob_val,
               (L > 0) ? sk_val : nullptr, XVALF, OUTVA);
    edge_stage(ei_vo, N_OPC, QOP, KRVO, VRVO, p_vo, ow_op, ob_op,
               (L > 0) ? sk_op : nullptr, XOPF, OUTOP);

    if (L < 2) {
      norm_stage(OUTOP, norm_w[0][L], norm_b[0][L], XOPF, XOPB);
      norm_stage(OUTVA, norm_w[1][L], norm_b[1][L], XVALF, XVALB);
    } else {
      k_gelu_f32<<<CDIV(NODE_ELEMS, 256), 256, 0, stream>>>(OUTOP, XOPF, NODE_ELEMS);
      k_gelu_f32<<<CDIV(NODE_ELEMS, 256), 256, 0, stream>>>(OUTVA, XVALF, NODE_ELEMS);
    }
  }

  // ---- pooling head ----
  const int EG = 2 * EDGES + NNODE;
  k_concat<<<CDIV(NNODE * 256, 256), 256, 0, stream>>>(XOPF, XVALF, XH, NNODE * 256);
  k_gat_node<<<CDIV(NNODE, 256), 256, 0, stream>>>(XH, gat_w, gat_as, gat_ad,
                                                   XP, AS_, AD_, GKEY, GSUM, SCORE);
  k_gat_logit<<<CDIV(EG, 256), 256, 0, stream>>>(ei_ov, ei_vo, AS_, AD_, ELOG, GKEY, EG);
  k_gat_exp<<<CDIV(EG, 256), 256, 0, stream>>>(ei_ov, ei_vo, ELOG, GKEY, GSUM, EG);
  k_gat_score<<<CDIV(EG, 256), 256, 0, stream>>>(ei_ov, ei_vo, ELOG, GSUM, XP, SCORE, EG);
  k_gat_tanh<<<CDIV(NNODE, 256), 256, 0, stream>>>(SCORE, gat_b);
  k_topk16<<<1, 1024, 0, stream>>>(SCORE, NNODE, TOPV, TOPI);
  k_head<<<1, 256, 0, stream>>>(XH, TOPV, TOPI, r1b, r1w, r2b, r2w,
                                m1b, m1w, m2b, m2w, m3b, m3w, m4b, m4w,
                                (float*)d_out);
}
